// Model_21706764714573
// MI455X (gfx1250) — matrix operand order repaired and checked by reading, not compiled, not run on hardware
//
#include <hip/hip_runtime.h>
#include <stdint.h>

// ---------------------------------------------------------------------------
// Problem dims (fixed by the reference)
// ---------------------------------------------------------------------------
#define B_SZ  128
#define S_LEN 512
#define I_DIM 512
#define H_DIM 1024

// ---------------------------------------------------------------------------
// WMMA vector types (CDNA5 / gfx1250, wave32)
// ---------------------------------------------------------------------------
typedef __attribute__((ext_vector_type(16))) __bf16 v16bf;
typedef __attribute__((ext_vector_type(8)))  float  v8f;

union V16U {
    v16bf          v;
    uint4          u[2];
    unsigned short s[16];
};

static __device__ __forceinline__ unsigned short f2bf(float f) {
    // round-to-nearest-even f32 -> bf16
    unsigned u = __float_as_uint(f);
    u += 0x7FFFu + ((u >> 16) & 1u);
    return (unsigned short)(u >> 16);
}
static __device__ __forceinline__ float bf2f(unsigned short s) {
    return __uint_as_float(((unsigned)s) << 16);
}

// ---------------------------------------------------------------------------
// Operand loaders (layouts per CDNA5 ISA 7.12.2).
//
// A operand (16x32 bf16, M x K): lane L holds row m = L&15, hi = L>>4:
//   halves 0..7  = K (k0 + hi*8 + 0..7)
//   halves 8..15 = K (k0 + hi*8 + 16..23)
// -> two contiguous 16-byte loads per lane from row-major [M][K] storage.
//
// B operand (32x16 bf16, K x N): lane L holds column n = n0 + (L&15), hi = L>>4:
//   halves 0..7  = K (k0 + hi*8 + 0..7)
//   halves 8..15 = K (k0 + hi*8 + 16..23)
// -> strided 16-bit loads per lane from PRE-TRANSPOSED weights
//    Wt[k][n] (row stride = 1024 = N).
// ---------------------------------------------------------------------------
static __device__ __forceinline__ v16bf load_b_tile(
        const unsigned short* __restrict__ Wt, int k0, int n0, int lane) {
    const unsigned short* p = Wt + (size_t)(k0 + (lane >> 4) * 8) * H_DIM + n0 + (lane & 15);
    V16U t;
#pragma unroll
    for (int i = 0; i < 8; ++i) {
        t.s[i]     = p[(size_t)i * H_DIM];
        t.s[i + 8] = p[(size_t)(i + 16) * H_DIM];
    }
    return t.v;
}

static __device__ __forceinline__ v16bf load_a_bf16(
        const unsigned short* __restrict__ A, size_t row_stride,
        int m, int k0, int hi) {
    const unsigned short* p = A + (size_t)m * row_stride + (size_t)(k0 + hi * 8);
    V16U t;
    t.u[0] = *(const uint4*)(p);
    t.u[1] = *(const uint4*)(p + 16);
    return t.v;
}

static __device__ __forceinline__ v16bf load_a_f32(
        const float* __restrict__ A, size_t row_stride, size_t off,
        int m, int k0, int hi) {
    const float* p = A + (size_t)m * row_stride + off + (size_t)(k0 + hi * 8);
    float4 f0 = *(const float4*)(p);
    float4 f1 = *(const float4*)(p + 4);
    float4 f2 = *(const float4*)(p + 16);
    float4 f3 = *(const float4*)(p + 20);
    V16U t;
    t.s[0]  = f2bf(f0.x); t.s[1]  = f2bf(f0.y); t.s[2]  = f2bf(f0.z); t.s[3]  = f2bf(f0.w);
    t.s[4]  = f2bf(f1.x); t.s[5]  = f2bf(f1.y); t.s[6]  = f2bf(f1.z); t.s[7]  = f2bf(f1.w);
    t.s[8]  = f2bf(f2.x); t.s[9]  = f2bf(f2.y); t.s[10] = f2bf(f2.z); t.s[11] = f2bf(f2.w);
    t.s[12] = f2bf(f3.x); t.s[13] = f2bf(f3.y); t.s[14] = f2bf(f3.z); t.s[15] = f2bf(f3.w);
    return t.v;
}

// Epilogue: +bias, tanh, store one 16x16 accumulator tile as bf16.
// C/D layout: VGPR r, lanes 0-15 -> (M=r, N=lane); lanes 16-31 -> (M=8+r).
static __device__ __forceinline__ void store_tile(
        unsigned short* __restrict__ out, const float* __restrict__ bias,
        v8f acc, int m0, int n0, int lane, int hi) {
    const int   n  = n0 + (lane & 15);
    const float bn = bias[n];
#pragma unroll
    for (int r = 0; r < 8; ++r) {
        const int m = m0 + hi * 8 + r;
        out[(size_t)m * H_DIM + n] = f2bf(tanhf(acc[r] + bn));
    }
}

// ---------------------------------------------------------------------------
// Diagonal step kernel: ONE launch computes layer-0 timestep s AND layer-1
// timestep s-1 (they are independent):
//   blocks  0..31 : h1[s]   = tanh(x[:,s,:] @ Wih0^T + h1[s-1] @ Whh0^T + b0)
//   blocks 32..63 : h2[s-1] = tanh(h1[s-1]  @ Wih1^T + h2[s-2] @ Whh1^T + b1)
// Each block: 8 waves over M=128 (16 rows/wave), 32-wide N strip; each wave
// computes a 16x32 tile (two v8f accumulators -> two independent WMMA chains
// per K-slab for latency hiding).
//
// XBF: layer-0 A operand comes from a pre-converted bf16 copy of x (fast
// path, no in-register cvt on the critical path); otherwise convert f32 x
// in-register (workspace-constrained fallback).
// ---------------------------------------------------------------------------
template <bool XBF>
__global__ __launch_bounds__(256)
void rnn_diag_kernel(const float* __restrict__ x,
                     const unsigned short* __restrict__ x_bf, int s,
                     const unsigned short* __restrict__ Wt_ih0, // [512][1024]
                     const unsigned short* __restrict__ Wt_hh0, // [1024][1024]
                     const float* __restrict__ bias0,
                     const unsigned short* __restrict__ h1_prev, // h1[s-1]
                     unsigned short* __restrict__ h1_cur,        // h1[s]
                     const unsigned short* __restrict__ Wt_ih1,
                     const unsigned short* __restrict__ Wt_hh1,
                     const float* __restrict__ bias1,
                     const unsigned short* __restrict__ h2_prev, // h2[s-2]
                     unsigned short* __restrict__ h2_cur)        // h2[s-1]
{
    const bool is_l1 = blockIdx.x >= 32;
    if (!is_l1 && s >= S_LEN) return;   // no layer-0 work on the drain launch
    if (is_l1 && s == 0) return;        // no layer-1 work on the fill launch

    const int lane = threadIdx.x & 31;
    const int wv   = threadIdx.x >> 5;
    const int m0   = wv * 16;                               // batch rows
    const int n0   = (is_l1 ? blockIdx.x - 32 : blockIdx.x) * 32; // hidden cols
    const int mrow = m0 + (lane & 15);
    const int hi   = lane >> 4;

    v8f acc0 = {0.f, 0.f, 0.f, 0.f, 0.f, 0.f, 0.f, 0.f};
    v8f acc1 = {0.f, 0.f, 0.f, 0.f, 0.f, 0.f, 0.f, 0.f};

    if (!is_l1) {
        // -------- layer 0, timestep s --------
        for (int k0 = 0; k0 < I_DIM; k0 += 32) {
            v16bf a;
            if (XBF) {
                a = load_a_bf16(x_bf + (size_t)s * I_DIM,
                                (size_t)S_LEN * I_DIM, mrow, k0, hi);
            } else {
                a = load_a_f32(x, (size_t)S_LEN * I_DIM, (size_t)s * I_DIM,
                               mrow, k0, hi);
            }
            v16bf b0 = load_b_tile(Wt_ih0, k0, n0, lane);
            v16bf b1 = load_b_tile(Wt_ih0, k0, n0 + 16, lane);
            acc0 = __builtin_amdgcn_wmma_f32_16x16x32_bf16(
                       false, a, false, b0, (short)0, acc0, false, false);
            acc1 = __builtin_amdgcn_wmma_f32_16x16x32_bf16(
                       false, a, false, b1, (short)0, acc1, false, false);
        }
        for (int k0 = 0; k0 < H_DIM; k0 += 32) {
            v16bf a  = load_a_bf16(h1_prev, H_DIM, mrow, k0, hi);
            v16bf b0 = load_b_tile(Wt_hh0, k0, n0, lane);
            v16bf b1 = load_b_tile(Wt_hh0, k0, n0 + 16, lane);
            acc0 = __builtin_amdgcn_wmma_f32_16x16x32_bf16(
                       false, a, false, b0, (short)0, acc0, false, false);
            acc1 = __builtin_amdgcn_wmma_f32_16x16x32_bf16(
                       false, a, false, b1, (short)0, acc1, false, false);
        }
        store_tile(h1_cur, bias0, acc0, m0, n0,      lane, hi);
        store_tile(h1_cur, bias0, acc1, m0, n0 + 16, lane, hi);
    } else {
        // -------- layer 1, timestep s-1 --------
        for (int k0 = 0; k0 < H_DIM; k0 += 32) {
            v16bf a  = load_a_bf16(h1_prev, H_DIM, mrow, k0, hi);
            v16bf b0 = load_b_tile(Wt_ih1, k0, n0, lane);
            v16bf b1 = load_b_tile(Wt_ih1, k0, n0 + 16, lane);
            acc0 = __builtin_amdgcn_wmma_f32_16x16x32_bf16(
                       false, a, false, b0, (short)0, acc0, false, false);
            acc1 = __builtin_amdgcn_wmma_f32_16x16x32_bf16(
                       false, a, false, b1, (short)0, acc1, false, false);
        }
        for (int k0 = 0; k0 < H_DIM; k0 += 32) {
            v16bf a  = load_a_bf16(h2_prev, H_DIM, mrow, k0, hi);
            v16bf b0 = load_b_tile(Wt_hh1, k0, n0, lane);
            v16bf b1 = load_b_tile(Wt_hh1, k0, n0 + 16, lane);
            acc0 = __builtin_amdgcn_wmma_f32_16x16x32_bf16(
                       false, a, false, b0, (short)0, acc0, false, false);
            acc1 = __builtin_amdgcn_wmma_f32_16x16x32_bf16(
                       false, a, false, b1, (short)0, acc1, false, false);
        }
        store_tile(h2_cur, bias1, acc0, m0, n0,      lane, hi);
        store_tile(h2_cur, bias1, acc1, m0, n0 + 16, lane, hi);
    }
}

// ---------------------------------------------------------------------------
// One-time prep kernels (off the critical path).
// ---------------------------------------------------------------------------

// f32 [N][K] -> bf16 transposed [K][N]
__global__ void transpose_cvt_kernel(const float* __restrict__ W,
                                     unsigned short* __restrict__ Wt,
                                     int N, int K) {
    int t = blockIdx.x * blockDim.x + threadIdx.x;
    if (t >= N * K) return;
    int k = t / N;
    int n = t - k * N;
    Wt[t] = f2bf(W[(size_t)n * K + k]);
}

// flat f32 -> bf16, 4 elements per thread (coalesced b128 in / b64 out)
__global__ void cvt_f32_bf16_kernel(const float* __restrict__ in,
                                    unsigned short* __restrict__ out, int n4) {
    int t = blockIdx.x * blockDim.x + threadIdx.x;
    if (t >= n4) return;
    float4 f = ((const float4*)in)[t];
    union { unsigned short s[4]; uint2 u; } o;
    o.s[0] = f2bf(f.x); o.s[1] = f2bf(f.y);
    o.s[2] = f2bf(f.z); o.s[3] = f2bf(f.w);
    ((uint2*)out)[t] = o.u;
}

__global__ void bias_sum_kernel(const float* __restrict__ a,
                                const float* __restrict__ b,
                                float* __restrict__ o, int n) {
    int t = blockIdx.x * blockDim.x + threadIdx.x;
    if (t < n) o[t] = a[t] + b[t];
}

// ---------------------------------------------------------------------------
// Head: logits = relu(h2_last @ W2^T + b2), [128,2].
// ---------------------------------------------------------------------------
__global__ __launch_bounds__(256)
void head_kernel(const unsigned short* __restrict__ h2,
                 const float* __restrict__ W2, const float* __restrict__ b2,
                 float* __restrict__ out) {
    const int t = threadIdx.x;           // 0..255
    const int b = t >> 1;
    const int j = t & 1;
    const unsigned short* hr = h2 + (size_t)b * H_DIM;
    const float*          wr = W2 + (size_t)j * H_DIM;
    float s = 0.f;
    for (int k = 0; k < H_DIM; ++k) s += bf2f(hr[k]) * wr[k];
    s += b2[j];
    out[b * 2 + j] = s > 0.f ? s : 0.f;
}

// ---------------------------------------------------------------------------
// Launch. Inputs (setup_inputs order):
// 0:x 1:mask 2:W_ih0 3:W_hh0 4:b_ih0 5:b_hh0 6:W_ih1 7:W_hh1 8:b_ih1 9:b_hh1
// 10:W2 11:b2
// ---------------------------------------------------------------------------
extern "C" void kernel_launch(void* const* d_in, const int* in_sizes, int n_in,
                              void* d_out, int out_size, void* d_ws, size_t ws_size,
                              hipStream_t stream) {
    (void)in_sizes; (void)n_in; (void)out_size;

    const float* x    = (const float*)d_in[0];
    const float* Wih0 = (const float*)d_in[2];
    const float* Whh0 = (const float*)d_in[3];
    const float* bih0 = (const float*)d_in[4];
    const float* bhh0 = (const float*)d_in[5];
    const float* Wih1 = (const float*)d_in[6];
    const float* Whh1 = (const float*)d_in[7];
    const float* bih1 = (const float*)d_in[8];
    const float* bhh1 = (const float*)d_in[9];
    const float* W2   = (const float*)d_in[10];
    const float* b2   = (const float*)d_in[11];

    // Workspace carve-out. Base ~8.4 MB (weights stay L2-resident) plus an
    // optional 64 MB bf16 copy of x if the workspace allows it.
    char*  ws  = (char*)d_ws;
    size_t off = 0;
    auto alloc = [&](size_t bytes) -> char* {
        char* p = ws + off;
        off += (bytes + 255) & ~(size_t)255;
        return p;
    };
    unsigned short* Wt_ih0 = (unsigned short*)alloc((size_t)I_DIM * H_DIM * 2);
    unsigned short* Wt_hh0 = (unsigned short*)alloc((size_t)H_DIM * H_DIM * 2);
    unsigned short* Wt_ih1 = (unsigned short*)alloc((size_t)H_DIM * H_DIM * 2);
    unsigned short* Wt_hh1 = (unsigned short*)alloc((size_t)H_DIM * H_DIM * 2);
    float*          bias0  = (float*)alloc((size_t)H_DIM * 4);
    float*          bias1  = (float*)alloc((size_t)H_DIM * 4);
    unsigned short* h1a    = (unsigned short*)alloc((size_t)B_SZ * H_DIM * 2);
    unsigned short* h1b    = (unsigned short*)alloc((size_t)B_SZ * H_DIM * 2);
    unsigned short* h2a    = (unsigned short*)alloc((size_t)B_SZ * H_DIM * 2);
    unsigned short* h2b    = (unsigned short*)alloc((size_t)B_SZ * H_DIM * 2);

    const size_t xbf_bytes = (size_t)B_SZ * S_LEN * I_DIM * 2;   // 64 MB
    const bool   use_xbf   = (ws_size >= off + xbf_bytes + 256);
    unsigned short* x_bf   = use_xbf ? (unsigned short*)alloc(xbf_bytes) : nullptr;

    // Weight conversion + transpose (bf16, [K][N]); fused biases.
    transpose_cvt_kernel<<<(I_DIM * H_DIM + 255) / 256, 256, 0, stream>>>(Wih0, Wt_ih0, H_DIM, I_DIM);
    transpose_cvt_kernel<<<(H_DIM * H_DIM + 255) / 256, 256, 0, stream>>>(Whh0, Wt_hh0, H_DIM, H_DIM);
    transpose_cvt_kernel<<<(H_DIM * H_DIM + 255) / 256, 256, 0, stream>>>(Wih1, Wt_ih1, H_DIM, H_DIM);
    transpose_cvt_kernel<<<(H_DIM * H_DIM + 255) / 256, 256, 0, stream>>>(Whh1, Wt_hh1, H_DIM, H_DIM);
    bias_sum_kernel<<<(H_DIM + 255) / 256, 256, 0, stream>>>(bih0, bhh0, bias0, H_DIM);
    bias_sum_kernel<<<(H_DIM + 255) / 256, 256, 0, stream>>>(bih1, bhh1, bias1, H_DIM);

    // Pre-convert x to bf16 once (removes per-step cvt VALU from the scan).
    if (use_xbf) {
        const int n4 = (B_SZ * S_LEN * I_DIM) / 4;
        cvt_f32_bf16_kernel<<<(n4 + 255) / 256, 256, 0, stream>>>(x, x_bf, n4);
    }

    // Zero initial states: h1[-1]=0, and both h2 buffers (h2[-1]=0 is first
    // read one launch AFTER the t=0 swap, so both ping-pong halves must be 0).
    hipMemsetAsync(h1a, 0, (size_t)B_SZ * H_DIM * 2, stream);
    hipMemsetAsync(h1b, 0, (size_t)B_SZ * H_DIM * 2, stream);
    hipMemsetAsync(h2a, 0, (size_t)B_SZ * H_DIM * 2, stream);
    hipMemsetAsync(h2b, 0, (size_t)B_SZ * H_DIM * 2, stream);

    // Diagonal-pipelined scan: 513 launches cover both layers (critical path).
    unsigned short *h1p = h1a, *h1c = h1b, *h2p = h2a, *h2c = h2b;
    for (int t = 0; t <= S_LEN; ++t) {
        if (use_xbf) {
            rnn_diag_kernel<true><<<64, 256, 0, stream>>>(
                x, x_bf, t,
                Wt_ih0, Wt_hh0, bias0, h1p, h1c,
                Wt_ih1, Wt_hh1, bias1, h2p, h2c);
        } else {
            rnn_diag_kernel<false><<<64, 256, 0, stream>>>(
                x, x_bf, t,
                Wt_ih0, Wt_hh0, bias0, h1p, h1c,
                Wt_ih1, Wt_hh1, bias1, h2p, h2c);
        }
        unsigned short* tmp;
        tmp = h1p; h1p = h1c; h1c = tmp;
        tmp = h2p; h2p = h2c; h2c = tmp;
    }

    // After the final swap h2p holds h2[S-1].
    head_kernel<<<1, 256, 0, stream>>>(h2p, W2, b2, (float*)d_out);
}
